// HSGMP_3246995275873
// MI455X (gfx1250) — matrix operand order repaired and checked by reading, not compiled, not run on hardware
//
#include <hip/hip_runtime.h>
#include <math.h>

// ---------------------------------------------------------------------------
// HSGMP forward for MI455X (gfx1250). Dense GEMMs run on WMMA f16->f32
// (v_wmma_f32_16x16x32_f16); LDS tiles laid out so every WMMA fragment is fed
// by ds_load_b128 (A: row-major runs, B: transposed tile). FAST GEMMs
// (lda%4==0 && K%32==0 -- all the big ones) have branch-free b128 staging;
// only the 300-wide input projections take the clamped scalar path.
// Irregular GAT edge ops use f32 global atomics.
// ---------------------------------------------------------------------------

typedef __attribute__((ext_vector_type(16))) _Float16 v16h;
typedef __attribute__((ext_vector_type(8)))  _Float16 v8h;
typedef __attribute__((ext_vector_type(8)))  float    v8f;

#define NN      50000
#define IMG_N   25000
#define EE      200000
#define HD_     512
#define COMMON_ 256
#define OUTF    256

#define LDS_STRIDE 40   // halves per row: 80B = 20 dwords -> bank-conflict-free,
                        // every fragment chunk stays 16B aligned

// ============================ WMMA GEMM ====================================
// C[M,N] = A[M,K] @ B[K,N] (+bias) (opt tanh). Row-major, N multiple of 64,
// ldb % 4 == 0. FAST: lda % 4 == 0 AND K % 32 == 0 (branch-free staging).
// Block: 128 threads (4 waves). Block tile 64x64; wave tile 16x64.
template<int ACT, int FAST>
__global__ __launch_bounds__(128)
void gemm_wmma_k(const float* __restrict__ A, int lda,
                 const float* __restrict__ B, int ldb,
                 const float* __restrict__ bias,
                 float* __restrict__ C, int ldc,
                 int M, int N, int K)
{
    __shared__ _Float16 As[64][LDS_STRIDE];   // [row][k]  64 rows x 32 K
    __shared__ _Float16 Bt[64][LDS_STRIDE];   // [col][k]  64 cols x 32 K (transposed)

    const int n0   = blockIdx.x * 64;
    const int m0   = blockIdx.y * 64;
    const int tid  = threadIdx.x;
    const int lane = tid & 31;
    const int wv   = tid >> 5;        // 0..3
    const int half = lane >> 4;       // 0/1 (lane group)
    const int l16  = lane & 15;

    v8f acc[4];
    #pragma unroll
    for (int t = 0; t < 4; ++t)
        #pragma unroll
        for (int j = 0; j < 8; ++j) acc[t][j] = 0.f;

    const int ar  = tid >> 1, ac0 = (tid & 1) * 16;   // A staging: half row each
    const int br  = tid >> 2, bc0 = (tid & 3) * 16;   // B staging: quarter row

    // Row clamp: rows >= M stage row M-1; their products land only in C rows
    // >= M which the epilogue never stores -> correct, and branch-free.
    const int arow_g = (m0 + ar < M) ? (m0 + ar) : (M - 1);
    const float* Ap = A + (long)arow_g * lda;

    for (int k0 = 0; k0 < K; k0 += 32) {
        if (FAST) {
            // -------- branch-free staging: global_load_b128 x4 each --------
            v8h t0, t1;
            {
                const float4* p = (const float4*)(Ap + k0 + ac0);
                float4 q0 = p[0], q1 = p[1], q2 = p[2], q3 = p[3];
                t0[0]=(_Float16)q0.x; t0[1]=(_Float16)q0.y; t0[2]=(_Float16)q0.z; t0[3]=(_Float16)q0.w;
                t0[4]=(_Float16)q1.x; t0[5]=(_Float16)q1.y; t0[6]=(_Float16)q1.z; t0[7]=(_Float16)q1.w;
                t1[0]=(_Float16)q2.x; t1[1]=(_Float16)q2.y; t1[2]=(_Float16)q2.z; t1[3]=(_Float16)q2.w;
                t1[4]=(_Float16)q3.x; t1[5]=(_Float16)q3.y; t1[6]=(_Float16)q3.z; t1[7]=(_Float16)q3.w;
            }
            *(v8h*)&As[ar][ac0]     = t0;
            *(v8h*)&As[ar][ac0 + 8] = t1;
            if (k0 + 32 < K)
                __builtin_prefetch(Ap + k0 + ac0 + 32, 0, 0);  // global_prefetch_b8

            // B: (k*ldb + n0 + bc0) always 16B aligned (ldb % 4 == 0)
            const float4* bp = (const float4*)(B + (long)(k0 + br) * ldb + n0 + bc0);
            float4 b0 = bp[0], b1 = bp[1], b2 = bp[2], b3 = bp[3];
            float bv[16] = { b0.x,b0.y,b0.z,b0.w, b1.x,b1.y,b1.z,b1.w,
                             b2.x,b2.y,b2.z,b2.w, b3.x,b3.y,b3.z,b3.w };
            #pragma unroll
            for (int i = 0; i < 16; ++i) Bt[bc0 + i][br] = (_Float16)bv[i];
            if (k0 + 32 < K)
                __builtin_prefetch((const float*)bp + (long)32 * ldb, 0, 0);
        } else {
            // ---- generic staging (K=300 projections): clamp + select ----
            v8h t0, t1;
            #pragma unroll
            for (int i = 0; i < 8; ++i) {
                int k = k0 + ac0 + i;  int kc = k < K ? k : K - 1;
                float v = Ap[kc];      t0[i] = (_Float16)(k < K ? v : 0.f);
            }
            #pragma unroll
            for (int i = 0; i < 8; ++i) {
                int k = k0 + ac0 + 8 + i;  int kc = k < K ? k : K - 1;
                float v = Ap[kc];          t1[i] = (_Float16)(k < K ? v : 0.f);
            }
            *(v8h*)&As[ar][ac0]     = t0;
            *(v8h*)&As[ar][ac0 + 8] = t1;

            const int gk  = k0 + br;
            const int gkc = gk < K ? gk : K - 1;
            const float* Bp = B + (long)gkc * ldb + n0 + bc0;
            #pragma unroll
            for (int i = 0; i < 16; ++i) {
                float v = Bp[i];
                Bt[bc0 + i][br] = (_Float16)(gk < K ? v : 0.f);
            }
        }
        __syncthreads();

        // ---- A fragment: 16x32 f16 per wave (rows wv*16..+15) ----
        // lanes 0-15: row=l16, K {0..7,16..23}; lanes 16-31: K {8..15,24..31}
        union { v16h v; v8h h[2]; } afr;
        const int arow = wv * 16 + l16;
        const int kb   = half * 8;
        afr.h[0] = *(const v8h*)&As[arow][kb];        // ds_load_b128
        afr.h[1] = *(const v8h*)&As[arow][kb + 16];   // ds_load_b128

        // ---- 4 WMMAs over the 64-wide N strip ----
        // B fragment: lane col=l16, K {0..7,16..23} / {8..15,24..31} (matches A) -> vector load
        #pragma unroll
        for (int t = 0; t < 4; ++t) {
            union { v16h v; v8h h[2]; } bfr;
            const _Float16* bp = &Bt[t * 16 + l16][half * 8];
            bfr.h[0] = *(const v8h*)bp;               // ds_load_b128
            bfr.h[1] = *(const v8h*)(bp + 16);        // ds_load_b128
            acc[t] = __builtin_amdgcn_wmma_f32_16x16x32_f16(
                false, afr.v, false, bfr.v, (short)0, acc[t], false, false);
        }
        __syncthreads();
    }

    // ---- epilogue: C/D layout = lane col l16, VGPR j -> row j + 8*half ----
    #pragma unroll
    for (int t = 0; t < 4; ++t) {
        const int gn = n0 + t * 16 + l16;
        const float bs = bias ? bias[gn] : 0.f;
        #pragma unroll
        for (int j = 0; j < 8; ++j) {
            const int gm = m0 + wv * 16 + half * 8 + j;
            if (gm < M) {
                float v = acc[t][j] + bs;
                if (ACT == 1) v = tanhf(v);
                C[(long)gm * ldc + gn] = v;
            }
        }
    }
}

// ============================ small kernels ================================
__global__ void fill_k(float* p, float v, long n) {
    long i = (long)blockIdx.x * blockDim.x + threadIdx.x;
    if (i < n) p[i] = v;
}

// el/er attention coefficients: el[n,h] = sum_d hproj[n,h*64+d]*al[h*64+d]
__global__ void attn_coef_k(const float* __restrict__ hp,
                            const float* __restrict__ al,
                            const float* __restrict__ ar,
                            float* __restrict__ el, float* __restrict__ er)
{
    int idx = blockIdx.x * blockDim.x + threadIdx.x;   // n*8+h
    if (idx >= NN * 8) return;
    int n = idx >> 3, h = idx & 7;
    const float* row = hp + (long)n * HD_ + h * 64;
    const float* a1 = al + h * 64;
    const float* a2 = ar + h * 64;
    float s1 = 0.f, s2 = 0.f;
    #pragma unroll 8
    for (int d = 0; d < 64; ++d) { float v = row[d]; s1 += v * a1[d]; s2 += v * a2[d]; }
    el[idx] = s1; er[idx] = s2;
}

__device__ inline void atomicMaxF(float* addr, float val) {
    if (val >= 0.f) atomicMax((int*)addr, __float_as_int(val));
    else            atomicMin((unsigned int*)addr, __float_as_uint(val));
}
__device__ inline float lrelu(float v) { return v > 0.f ? v : 0.2f * v; }

__global__ void edge_max_k(const int* __restrict__ src, const int* __restrict__ dst,
                           const float* __restrict__ el, const float* __restrict__ er,
                           float* __restrict__ emax)
{
    int idx = blockIdx.x * blockDim.x + threadIdx.x;   // e*8+h
    if (idx >= EE * 8) return;
    int e = idx >> 3, h = idx & 7;
    int s = src[e], d = dst[e];
    atomicMaxF(&emax[d * 8 + h], lrelu(el[s * 8 + h] + er[d * 8 + h]));
}

__global__ void emax_fix_k(float* emax) {             // empty segments -> 0
    int i = blockIdx.x * blockDim.x + threadIdx.x;
    if (i < NN * 8 && emax[i] < -1e29f) emax[i] = 0.f;
}

__global__ void edge_exp_k(const int* __restrict__ src, const int* __restrict__ dst,
                           const float* __restrict__ el, const float* __restrict__ er,
                           const float* __restrict__ emax,
                           float* __restrict__ exw, float* __restrict__ denom)
{
    int idx = blockIdx.x * blockDim.x + threadIdx.x;
    if (idx >= EE * 8) return;
    int e = idx >> 3, h = idx & 7;
    int s = src[e], d = dst[e];
    float x = __expf(lrelu(el[s * 8 + h] + er[d * 8 + h]) - emax[d * 8 + h]);
    exw[idx] = x;
    atomicAdd(&denom[d * 8 + h], x);
}

// num[dst,:] += exw[e,h] * hproj[src,:]   (block per edge, 256 threads)
__global__ __launch_bounds__(256)
void edge_aggr_k(const int* __restrict__ src, const int* __restrict__ dst,
                 const float* __restrict__ exw, const float* __restrict__ hp,
                 float* __restrict__ num)
{
    __shared__ float w[8];
    int e = blockIdx.x;
    if (threadIdx.x < 8) w[threadIdx.x] = exw[e * 8 + threadIdx.x];
    __syncthreads();
    int s = src[e], d = dst[e];
    const float* hs = hp + (long)s * HD_;
    float* nd = num + (long)d * HD_;
    for (int f = threadIdx.x; f < HD_; f += 256)
        atomicAdd(&nd[f], w[f >> 6] * hs[f]);
}

__global__ void gat_final_k(float* __restrict__ num, const float* __restrict__ denom,
                            const float* __restrict__ b)
{
    long idx = (long)blockIdx.x * blockDim.x + threadIdx.x;
    if (idx >= (long)NN * HD_) return;
    int n = (int)(idx >> 9), f = (int)(idx & 511);
    float v = num[idx] / fmaxf(denom[n * 8 + (f >> 6)], 1e-9f) + b[f];
    num[idx] = v > 0.f ? v : __expf(v) - 1.f;          // elu, in place
}

// semantic attention: wave per row, dot(tanh-projected row, w2) -> acc[m]
__global__ void sem_rowdot_k(const float* __restrict__ T, const float* __restrict__ w2,
                             float* __restrict__ acc)
{
    int row  = (blockIdx.x * blockDim.x + threadIdx.x) >> 5;
    int lane = threadIdx.x & 31;
    if (row >= NN) return;
    float s = 0.f;
    for (int j = lane; j < 128; j += 32) s += T[(long)row * 128 + j] * w2[j];
    #pragma unroll
    for (int m = 16; m; m >>= 1) s += __shfl_xor(s, m, 32);
    if (!lane) atomicAdd(acc, s);
}

__global__ void sem_combine_k(const float* __restrict__ zA, const float* __restrict__ zB,
                              const float* __restrict__ acc, float* __restrict__ out,
                              float invN)
{
    long idx = (long)blockIdx.x * blockDim.x + threadIdx.x;
    if (idx >= (long)NN * HD_) return;
    float w0 = acc[0] * invN, w1 = acc[1] * invN;
    float m = fmaxf(w0, w1);
    float e0 = __expf(w0 - m), e1 = __expf(w1 - m);
    float r = 1.f / (e0 + e1);
    out[idx] = (e0 * r) * zA[idx] + (e1 * r) * zB[idx];
}

// column reduce: out[f] = scale * sum_n (wt? wt[n]:1) * Mtx[n*ld+f]; grid = ld
__global__ __launch_bounds__(256)
void col_reduce_k(const float* __restrict__ Mtx, const float* __restrict__ wt,
                  float* __restrict__ out, int rows, int ld, float scale)
{
    __shared__ float sh[256];
    int f = blockIdx.x;
    float s = 0.f;
    for (int n = threadIdx.x; n < rows; n += 256) {
        float v = Mtx[(long)n * ld + f];
        s += wt ? wt[n] * v : v;
    }
    sh[threadIdx.x] = s; __syncthreads();
    for (int st = 128; st; st >>= 1) {
        if (threadIdx.x < st) sh[threadIdx.x] += sh[threadIdx.x + st];
        __syncthreads();
    }
    if (!threadIdx.x) out[f] = sh[0] * scale;
}

// c = tanh(vin @ W + b), 256->256, single block of 256 threads
__global__ __launch_bounds__(256)
void vecmat_tanh_k(const float* __restrict__ vin, const float* __restrict__ W,
                   const float* __restrict__ b, float* __restrict__ out)
{
    __shared__ float sv[256];
    sv[threadIdx.x] = vin[threadIdx.x]; __syncthreads();
    float s = b[threadIdx.x];
    for (int i = 0; i < 256; ++i) s += sv[i] * W[i * 256 + threadIdx.x];
    out[threadIdx.x] = tanhf(s);
}

// gate[n] = sigmoid(pred[n,:] . c)   (wave per row)
__global__ void row_gate_k(const float* __restrict__ pred, const float* __restrict__ c,
                           float* __restrict__ gate, int rows)
{
    int row  = (blockIdx.x * blockDim.x + threadIdx.x) >> 5;
    int lane = threadIdx.x & 31;
    if (row >= rows) return;
    float s = 0.f;
    for (int f = lane; f < 256; f += 32) s += pred[(long)row * 256 + f] * c[f];
    #pragma unroll
    for (int m = 16; m; m >>= 1) s += __shfl_xor(s, m, 32);
    if (!lane) gate[row] = 1.f / (1.f + __expf(-s));
}

// NTN bilinear + tiny MLP tail -> scalar. Single block, 256 threads.
__global__ __launch_bounds__(256)
void ntn_mlp_k(const float* __restrict__ gv, const float* __restrict__ gt,
               const float* __restrict__ ntnW, const float* __restrict__ ntnU,
               const float* __restrict__ ntnUb,
               const float* __restrict__ s1W, const float* __restrict__ s1b,
               const float* __restrict__ s2W, const float* __restrict__ s2b,
               const float* __restrict__ s3W, const float* __restrict__ s3b,
               float* __restrict__ out)
{
    __shared__ float red[256];
    __shared__ float sgv[256], sgt[256];
    __shared__ float x[16];
    sgv[threadIdx.x] = gv[threadIdx.x];
    sgt[threadIdx.x] = gt[threadIdx.x];
    __syncthreads();
    for (int k = 0; k < 16; ++k) {
        const float* wrow = ntnW + ((long)k * 256 + threadIdx.x) * 256;
        float d = 0.f;
        for (int j = 0; j < 256; ++j) d += wrow[j] * sgt[j];
        red[threadIdx.x] = sgv[threadIdx.x] * d;
        __syncthreads();
        for (int st = 128; st; st >>= 1) {
            if (threadIdx.x < st) red[threadIdx.x] += red[threadIdx.x + st];
            __syncthreads();
        }
        if (!threadIdx.x) x[k] = red[0];
        __syncthreads();
    }
    if (!threadIdx.x) {
        float xv[16];
        for (int k = 0; k < 16; ++k) {
            float u = ntnUb[k];
            for (int i = 0; i < 256; ++i) u += sgv[i] * ntnU[i * 16 + k];
            for (int i = 0; i < 256; ++i) u += sgt[i] * ntnU[(256 + i) * 16 + k];
            xv[k] = tanhf(x[k] + u);
        }
        float a1[8];
        for (int j = 0; j < 8; ++j) {
            float s = s1b[j];
            for (int k = 0; k < 16; ++k) s += xv[k] * s1W[k * 8 + j];
            a1[j] = fmaxf(s, 0.f);
        }
        float a2[4];
        for (int j = 0; j < 4; ++j) {
            float s = s2b[j];
            for (int k = 0; k < 8; ++k) s += a1[k] * s2W[k * 4 + j];
            a2[j] = fmaxf(s, 0.f);
        }
        float o = s3b[0];
        for (int k = 0; k < 4; ++k) o += a2[k] * s3W[k];
        out[0] = o;
    }
}

// ============================ host orchestration ===========================
static void gemm(const float* A, int lda, const float* B, int ldb, const float* bias,
                 float* C, int ldc, int M, int N, int K, int act, hipStream_t s)
{
    dim3 grid(N / 64, (M + 63) / 64), blk(128);
    const bool fast = ((lda % 4) == 0) && ((K % 32) == 0);
    if (act == 1) {
        // tanh GEMMs always have lda=512, K=512 -> fast
        gemm_wmma_k<1,1><<<grid, blk, 0, s>>>(A, lda, B, ldb, bias, C, ldc, M, N, K);
    } else if (fast) {
        gemm_wmma_k<0,1><<<grid, blk, 0, s>>>(A, lda, B, ldb, bias, C, ldc, M, N, K);
    } else {
        gemm_wmma_k<0,0><<<grid, blk, 0, s>>>(A, lda, B, ldb, bias, C, ldc, M, N, K);
    }
}
static void fill(float* p, float v, long n, hipStream_t s) {
    fill_k<<<(int)((n + 255) / 256), 256, 0, s>>>(p, v, n);
}

static void run_gat(const float* X, int K, const float* W, const float* al,
                    const float* ar, const float* b, const int* src, const int* dst,
                    float* hproj, float* el, float* er, float* emax, float* denom,
                    float* exw, float* G, hipStream_t s)
{
    gemm(X, K, W, HD_, nullptr, hproj, HD_, NN, HD_, K, 0, s);          // h = X @ W
    attn_coef_k<<<(NN * 8 + 255) / 256, 256, 0, s>>>(hproj, al, ar, el, er);
    fill(emax, -1e30f, (long)NN * 8, s);
    fill(denom, 0.f, (long)NN * 8, s);
    fill(G, 0.f, (long)NN * HD_, s);
    edge_max_k<<<(EE * 8 + 255) / 256, 256, 0, s>>>(src, dst, el, er, emax);
    emax_fix_k<<<(NN * 8 + 255) / 256, 256, 0, s>>>(emax);
    edge_exp_k<<<(EE * 8 + 255) / 256, 256, 0, s>>>(src, dst, el, er, emax, exw, denom);
    edge_aggr_k<<<EE, 256, 0, s>>>(src, dst, exw, hproj, G);
    gat_final_k<<<(int)(((long)NN * HD_ + 255) / 256), 256, 0, s>>>(G, denom, b);
}

extern "C" void kernel_launch(void* const* d_in, const int* in_sizes, int n_in,
                              void* d_out, int out_size, void* d_ws, size_t ws_size,
                              hipStream_t stream)
{
    (void)in_sizes; (void)n_in; (void)out_size; (void)ws_size;
    // tensors
    const float* img_obj = (const float*)d_in[0];   // [10000,1024]
    const float* img_rel = (const float*)d_in[1];   // [15000,1024]
    const float* txt_obj = (const float*)d_in[2];   // [15000,300]
    const float* txt_rel = (const float*)d_in[3];   // [10000,300]
    const int* e_src[2] = { (const int*)d_in[4], (const int*)d_in[6] };
    const int* e_dst[2] = { (const int*)d_in[5], (const int*)d_in[7] };
    // params, jax pytree order (dict keys sorted; 'W' < 'al' in ASCII)
    const float* fc_o_W = (const float*)d_in[8];
    const float* fc_o_b = (const float*)d_in[9];
    const float* fc_p_W = (const float*)d_in[10];
    const float* fc_p_b = (const float*)d_in[11];
    const float* fc_r_W = (const float*)d_in[12];
    const float* fc_r_b = (const float*)d_in[13];
    const float* fc_w_W = (const float*)d_in[14];
    const float* fc_w_b = (const float*)d_in[15];
    const float* gW[2][2], *gal[2][2], *gar[2][2], *gb[2][2];
    const float* semW1[2], *semb1[2], *semw2[2];
    for (int l = 0; l < 2; ++l) {
        int base = (l == 0) ? 16 : 27;
        for (int g = 0; g < 2; ++g) {
            gW[l][g]  = (const float*)d_in[base + g * 4 + 0];
            gal[l][g] = (const float*)d_in[base + g * 4 + 1];
            gar[l][g] = (const float*)d_in[base + g * 4 + 2];
            gb[l][g]  = (const float*)d_in[base + g * 4 + 3];
        }
        semW1[l] = (const float*)d_in[base + 8];
        semb1[l] = (const float*)d_in[base + 9];
        semw2[l] = (const float*)d_in[base + 10];
    }
    const float* ntn_U  = (const float*)d_in[38];
    const float* ntn_Ub = (const float*)d_in[39];
    const float* ntn_W  = (const float*)d_in[40];
    const float* pred_W = (const float*)d_in[41];
    const float* pred_b = (const float*)d_in[42];
    const float* s1_W = (const float*)d_in[43]; const float* s1_b = (const float*)d_in[44];
    const float* s2_W = (const float*)d_in[45]; const float* s2_b = (const float*)d_in[46];
    const float* s3_W = (const float*)d_in[47]; const float* s3_b = (const float*)d_in[48];
    const float* tc_W = (const float*)d_in[49]; const float* tc_b = (const float*)d_in[50];
    const float* vc_W = (const float*)d_in[51]; const float* vc_b = (const float*)d_in[52];

    // workspace bump allocator (~475 MB fp32; buffers aggressively reused)
    float* ws = (float*)d_ws;
    size_t off = 0;
    auto alloc = [&](long n) { float* p = ws + off; off += (size_t)n; return p; };
    float* h0    = alloc((long)NN * COMMON_);  // layer-0 input; later: pred
    float* gatA  = alloc((long)NN * HD_);
    float* gatB  = alloc((long)NN * HD_);
    float* hproj = alloc((long)NN * HD_);      // also reused as sem projection buf
    float* lyr   = alloc((long)NN * HD_);      // layer output (in-place for l1)
    float* el    = alloc((long)NN * 8);
    float* er    = alloc((long)NN * 8);
    float* emax  = alloc((long)NN * 8);
    float* denom = alloc((long)NN * 8);
    float* exw   = alloc((long)EE * 8);
    float* sem_acc = alloc(2);
    float* mean_v = alloc(256); float* mean_t = alloc(256);
    float* c_v = alloc(256);    float* c_t = alloc(256);
    float* g_v = alloc(256);    float* g_t = alloc(256);
    float* gate = alloc(IMG_N);

    // ---- input projections -> h0 [N,256] (concat order: o, r, w, p) ----
    gemm(img_obj, 1024, fc_o_W, COMMON_, fc_o_b, h0,                      COMMON_, 10000, COMMON_, 1024, 0, stream);
    gemm(img_rel, 1024, fc_r_W, COMMON_, fc_r_b, h0 + (long)10000*COMMON_,COMMON_, 15000, COMMON_, 1024, 0, stream);
    gemm(txt_obj,  300, fc_w_W, COMMON_, fc_w_b, h0 + (long)25000*COMMON_,COMMON_, 15000, COMMON_,  300, 0, stream);
    gemm(txt_rel,  300, fc_p_W, COMMON_, fc_p_b, h0 + (long)40000*COMMON_,COMMON_, 10000, COMMON_,  300, 0, stream);

    // ---- two HMP layers ----
    const float* lin = h0; int lK = COMMON_;
    for (int l = 0; l < 2; ++l) {
        run_gat(lin, lK, gW[l][0], gal[l][0], gar[l][0], gb[l][0],
                e_src[0], e_dst[0], hproj, el, er, emax, denom, exw, gatA, stream);
        run_gat(lin, lK, gW[l][1], gal[l][1], gar[l][1], gb[l][1],
                e_src[1], e_dst[1], hproj, el, er, emax, denom, exw, gatB, stream);
        // semantic attention (hproj region reused for the 128-wide projection)
        fill(sem_acc, 0.f, 2, stream);
        gemm(gatA, HD_, semW1[l], 128, semb1[l], hproj, 128, NN, 128, HD_, 1, stream);
        sem_rowdot_k<<<(NN * 32 + 255) / 256, 256, 0, stream>>>(hproj, semw2[l], &sem_acc[0]);
        gemm(gatB, HD_, semW1[l], 128, semb1[l], hproj, 128, NN, 128, HD_, 1, stream);
        sem_rowdot_k<<<(NN * 32 + 255) / 256, 256, 0, stream>>>(hproj, semw2[l], &sem_acc[1]);
        sem_combine_k<<<(int)(((long)NN * HD_ + 255) / 256), 256, 0, stream>>>(
            gatA, gatB, sem_acc, lyr, 1.f / (float)NN);
        lin = lyr; lK = HD_;
    }

    // ---- pred = lyr @ pred_W + b  (reuse h0 region) ----
    float* pred = h0;
    gemm(lyr, HD_, pred_W, OUTF, pred_b, pred, OUTF, NN, OUTF, HD_, 0, stream);

    // ---- context gates ----
    const float* pred_t = pred + (long)IMG_N * OUTF;
    col_reduce_k<<<256, 256, 0, stream>>>(pred,   nullptr, mean_v, IMG_N, OUTF, 1.f / IMG_N);
    col_reduce_k<<<256, 256, 0, stream>>>(pred_t, nullptr, mean_t, IMG_N, OUTF, 1.f / IMG_N);
    vecmat_tanh_k<<<1, 256, 0, stream>>>(mean_v, vc_W, vc_b, c_v);
    vecmat_tanh_k<<<1, 256, 0, stream>>>(mean_t, tc_W, tc_b, c_t);
    row_gate_k<<<(IMG_N * 32 + 255) / 256, 256, 0, stream>>>(pred, c_v, gate, IMG_N);
    col_reduce_k<<<256, 256, 0, stream>>>(pred,   gate, g_v, IMG_N, OUTF, 1.f);
    row_gate_k<<<(IMG_N * 32 + 255) / 256, 256, 0, stream>>>(pred_t, c_t, gate, IMG_N);
    col_reduce_k<<<256, 256, 0, stream>>>(pred_t, gate, g_t, IMG_N, OUTF, 1.f);

    // ---- NTN bilinear + MLP -> scalar ----
    ntn_mlp_k<<<1, 256, 0, stream>>>(g_v, g_t, ntn_W, ntn_U, ntn_Ub,
                                     s1_W, s1_b, s2_W, s2_b, s3_W, s3_b,
                                     (float*)d_out);
}
